// SSIM_36730560315957
// MI455X (gfx1250) — matrix operand order repaired and checked by reading, not compiled, not run on hardware
//
#include <hip/hip_runtime.h>
#include <hip/hip_bf16.h>

typedef __attribute__((ext_vector_type(16))) __bf16 v16bf;
typedef __attribute__((ext_vector_type(8)))  float  v8f;

union U16Frag { uint4 q[2]; v16bf v; };

// ---------------- constants ----------------
#define S_LEN   176400
#define N_FFT   2048
#define HOP     441
#define T_FRM   401      // frames
#define F_BIN   1025     // rfft bins
#define N_COLS  2050     // re/im interleaved
#define N_PAD   2064     // padded to multiple of 16
#define M_TILES 26       // ceil(401/16)
#define N_TILES 129      // ceil(2064/16)
#define A_STRIDE 2056    // 2048 + 8 bf16 pad (bank-conflict-free b128 reads)
#define IMG     64       // B*C
#define CROP_T  395
#define CROP_F  1019

__device__ __forceinline__ unsigned short f32_to_bf16(float f) {
    unsigned int u = __float_as_uint(f);
    u = (u + 0x7FFFu + ((u >> 16) & 1u)) >> 16;   // round-to-nearest-even
    return (unsigned short)u;
}

// ---------------- 1) bf16 DFT table, N-major: wt[col][k] ----------------
__global__ void dft_table_kernel(unsigned short* __restrict__ wt) {
    int idx = blockIdx.x * 256 + threadIdx.x;          // over N_PAD*2048
    if (idx >= N_PAD * N_FFT) return;
    int c = idx >> 11;          // column 0..2063
    int k = idx & 2047;
    float val = 0.0f;
    if (c < N_COLS) {
        int f  = c >> 1;
        int ph = (int)(((long long)f * (long long)k) & 2047);   // exact mod-2048 phase
        float ang = (float)ph * 3.0679615757712823e-3f;         // 2*pi/2048
        val = (c & 1) ? -__sinf(ang) : __cosf(ang);             // rfft: re=cos, im=-sin
    }
    wt[idx] = f32_to_bf16(val);
}

// ---------------- 2) windowed-frame x DFT GEMM via WMMA + magnitude ----------------
__global__ __launch_bounds__(256) void spec_gemm_kernel(
    const float* __restrict__ wavO, const float* __restrict__ wavT,
    const unsigned short* __restrict__ wt,
    float* __restrict__ specO, float* __restrict__ specT)
{
    extern __shared__ char smem[];
    unsigned short* sA = (unsigned short*)smem;        // [16][A_STRIDE] bf16

    const int tid = threadIdx.x;
    const int img = blockIdx.y;                        // 0..127 (0..63 output, 64..127 target)
    const int m0  = blockIdx.x * 16;                   // frame-tile base
    const int bc  = img & 63;
    const float* wav = (img < IMG) ? wavO : wavT;
    float*       spec = (img < IMG) ? specO : specT;
    wav  += (size_t)bc * S_LEN;
    spec += (size_t)bc * ((size_t)T_FRM * F_BIN);

    // stage 16 Hann-windowed frames as bf16 into LDS (reflect padding)
    for (int i = 0; i < 128; ++i) {
        int idx = tid + i * 256;                       // 16*2048 elements
        int m = idx >> 11, k = idx & 2047;
        int frame = m0 + m;
        float v = 0.0f;
        if (frame < T_FRM) {
            int p = frame * HOP + k - 1024;            // reflect into [0,S)
            if (p < 0) p = -p; else if (p >= S_LEN) p = 2 * S_LEN - 2 - p;
            float h = 0.5f - 0.5f * __cosf((float)k * 3.0679615757712823e-3f);
            v = wav[p] * h;
        }
        sA[m * A_STRIDE + k] = f32_to_bf16(v);
    }
    __syncthreads();

    const int lane = tid & 31;
    const int wv   = tid >> 5;          // wave id 0..7
    const int half = lane >> 4;         // 0: lanes 0-15, 1: lanes 16-31
    const int lm   = lane & 15;

    const unsigned short* arow = sA + lm * A_STRIDE + half * 8;

    for (int tile = wv; tile < N_TILES; tile += 8) {
        const int col0 = tile * 16;
        const unsigned short* bcol = wt + (size_t)(col0 + lm) * N_FFT + half * 8;

        v8f acc = {};
        for (int k0 = 0; k0 < N_FFT; k0 += 32) {
            U16Frag a, b;
            const uint4* ap = (const uint4*)(arow + k0);
            a.q[0] = ap[0];                 // K: k0+half*8 .. +7
            a.q[1] = ap[2];                 // K: k0+16+half*8 .. +7  (+32 bytes)
            const uint4* bp = (const uint4*)(bcol + k0);
            b.q[0] = bp[0];                 // K: k0+half*8 .. +7
            b.q[1] = bp[2];                 // K: k0+16+half*8 .. +7  (+32 bytes)
            acc = __builtin_amdgcn_wmma_f32_16x16x32_bf16(
                false, a.v, false, b.v, (short)0, acc, false, false);
        }

        // pair (re,im) across adjacent lanes, store magnitude
        #pragma unroll
        for (int r = 0; r < 8; ++r) {
            float v = acc[r];
            float o = __shfl_xor(v, 1, 32);
            float mag = sqrtf(fmaxf(v * v + o * o, 1e-8f));
            int frame = m0 + r + half * 8;
            int colc  = col0 + lm;
            if (((lm & 1) == 0) && frame < T_FRM && colc < N_COLS)
                spec[(size_t)frame * F_BIN + (colc >> 1)] = mag;
        }
    }
}

// ---------------- 3) SSIM: one block per (image, cropped row) ----------------
__global__ __launch_bounds__(256) void ssim_row_kernel(
    const float* __restrict__ specO, const float* __restrict__ specT,
    float* __restrict__ rowsum)
{
    __shared__ float sX[7 * F_BIN];
    __shared__ float sY[7 * F_BIN];
    __shared__ float red[256];

    const int tid = threadIdx.x;
    const int img = blockIdx.y;        // 0..63
    const int row = blockIdx.x;        // 0..394 ; window rows row..row+6
    const float* X = specO + (size_t)img * T_FRM * F_BIN + (size_t)row * F_BIN;
    const float* Y = specT + (size_t)img * T_FRM * F_BIN + (size_t)row * F_BIN;

    for (int i = tid; i < 7 * F_BIN; i += 256) { sX[i] = X[i]; sY[i] = Y[i]; }
    __syncthreads();

    const float inv49 = 1.0f / 49.0f;
    const float covn  = 49.0f / 48.0f;
    const float C1 = 4.0e-4f;          // (0.01*2)^2
    const float C2 = 3.6e-3f;          // (0.03*2)^2

    float local = 0.0f;
    for (int f = 3 + tid; f <= 1021; f += 256) {
        float sx = 0, sy = 0, sxx = 0, syy = 0, sxy = 0;
        #pragma unroll
        for (int dy = 0; dy < 7; ++dy) {
            int base = dy * F_BIN + f - 3;
            #pragma unroll
            for (int dx = 0; dx < 7; ++dx) {
                float x = sX[base + dx], y = sY[base + dx];
                sx += x; sy += y; sxx += x * x; syy += y * y; sxy += x * y;
            }
        }
        float ux = sx * inv49, uy = sy * inv49;
        float vx  = covn * (sxx * inv49 - ux * ux);
        float vy  = covn * (syy * inv49 - uy * uy);
        float vxy = covn * (sxy * inv49 - ux * uy);
        float A1 = 2.0f * ux * uy + C1, A2 = 2.0f * vxy + C2;
        float B1 = ux * ux + uy * uy + C1, B2 = vx + vy + C2;
        local += (A1 * A2) / (B1 * B2);
    }

    red[tid] = local;
    __syncthreads();
    for (int s = 128; s > 0; s >>= 1) {
        if (tid < s) red[tid] += red[tid + s];
        __syncthreads();
    }
    if (tid == 0) rowsum[img * CROP_T + row] = red[0];
}

// ---------------- 4) final reduction -> d_out[64] ----------------
__global__ void ssim_final_kernel(const float* __restrict__ rowsum, float* __restrict__ out) {
    int i = threadIdx.x;
    if (i < IMG) {
        float s = 0.0f;
        for (int r = 0; r < CROP_T; ++r) s += rowsum[i * CROP_T + r];
        out[i] = s / ((float)CROP_T * (float)CROP_F);
    }
}

// ---------------- host launcher ----------------
extern "C" void kernel_launch(void* const* d_in, const int* in_sizes, int n_in,
                              void* d_out, int out_size, void* d_ws, size_t ws_size,
                              hipStream_t stream) {
    (void)in_sizes; (void)n_in; (void)out_size; (void)ws_size;
    const float* wavO = (const float*)d_in[0];
    const float* wavT = (const float*)d_in[1];
    float* out = (float*)d_out;

    char* ws = (char*)d_ws;
    size_t off = 0;
    unsigned short* wt = (unsigned short*)(ws + off);
    off += (size_t)N_PAD * N_FFT * 2;                       // 8,454,144 B
    float* specO = (float*)(ws + off);
    off += (((size_t)IMG * T_FRM * F_BIN * 4) + 255) & ~(size_t)255;  // 105,222,400 B
    float* specT = (float*)(ws + off);
    off += (((size_t)IMG * T_FRM * F_BIN * 4) + 255) & ~(size_t)255;
    float* rowsum = (float*)(ws + off);                     // 64*395 floats

    // 1) DFT table
    dft_table_kernel<<<(N_PAD * N_FFT + 255) / 256, 256, 0, stream>>>(wt);

    // 2) WMMA spectrogram GEMM (dynamic LDS: 16*A_STRIDE bf16 = 65,792 B)
    spec_gemm_kernel<<<dim3(M_TILES, 128), 256, 16 * A_STRIDE * 2, stream>>>(
        wavO, wavT, wt, specO, specT);

    // 3) SSIM per cropped row
    ssim_row_kernel<<<dim3(CROP_T, IMG), 256, 0, stream>>>(specO, specT, rowsum);

    // 4) final means
    ssim_final_kernel<<<1, 64, 0, stream>>>(rowsum, out);
}
